// SparseFactorisationDense_70145405878819
// MI455X (gfx1250) — compile-verified, matrix operand order checked by reading, not run on hardware
//
#include <hip/hip_runtime.h>
#include <hip/hip_bf16.h>

// ---------------------------------------------------------------------------
// out = (S3 @ Bd2^T @ S2 @ Bd1^T @ S1 @ Bd0^T @ S0 @ x^T)^T + bias
// Batch-major:  Y <- Y @ W_k  with W0 = softmax(P0),
//               Wk = blockdiag(K_{k-1}) @ softmax(Pk)^T (folded into softmax).
// => 4 chained GEMMs [32768x1024]@[1024x1024] on v_wmma_f32_16x16x32_bf16,
//    A/B tiles DMA'd via GLOBAL_LOAD_ASYNC_TO_LDS_B128 (ASYNCcnt).
// ---------------------------------------------------------------------------

typedef __bf16 bf16_t;
typedef __attribute__((ext_vector_type(16))) __bf16 bf16x16;
typedef __attribute__((ext_vector_type(8)))  float  v8f;
typedef int v4i __attribute__((vector_size(16)));   // matches builtin param type

#define D_DIM 1024
#define B_DIM 32768
#define BM 128                    // block tile M
#define BN 256                    // block tile N
#define BK 32                     // K step
#define LDS_LD (BK + 8)           // padded LDS row stride (elements)
#define NK (D_DIM / BK)           // 32 K-steps

#if defined(__AMDGCN__) && __has_builtin(__builtin_amdgcn_global_load_async_to_lds_b128)
#define USE_ASYNC 1
#else
#define USE_ASYNC 0
#endif

union Frag16 {
    uint4   u[2];
    bf16_t  h[16];
    bf16x16 v;
};

// 16-byte global -> LDS copy. Async DMA on gfx1250 (no VGPR staging).
__device__ __forceinline__ void copy16_g2l(const bf16_t* g, bf16_t* l)
{
#if USE_ASYNC
    __builtin_amdgcn_global_load_async_to_lds_b128(
        (__attribute__((address_space(1))) v4i*)(uintptr_t)g,
        (__attribute__((address_space(3))) v4i*)(uint32_t)(uintptr_t)l,
        0, 0);
#else
    *(uint4*)l = *(const uint4*)g;
#endif
}

__device__ __forceinline__ void wait_async_zero()
{
#if USE_ASYNC
#if __has_builtin(__builtin_amdgcn_s_wait_asynccnt)
    __builtin_amdgcn_s_wait_asynccnt(0);
#else
    asm volatile("s_wait_asynccnt 0x0" ::: "memory");
#endif
#endif
}

// ---------------------------------------------------------------------------
// x fp32 -> bf16 (row-major copy, 8 elem/thread, fully coalesced)
// ---------------------------------------------------------------------------
__global__ __launch_bounds__(256)
void convert_bf16_kernel(const float* __restrict__ X, bf16_t* __restrict__ Y)
{
    const size_t i = ((size_t)blockIdx.x * 256 + threadIdx.x) * 8;
    const float4 f0 = *(const float4*)(X + i);
    const float4 f1 = *(const float4*)(X + i + 4);
    Frag16 h;
    h.h[0] = (bf16_t)f0.x; h.h[1] = (bf16_t)f0.y;
    h.h[2] = (bf16_t)f0.z; h.h[3] = (bf16_t)f0.w;
    h.h[4] = (bf16_t)f1.x; h.h[5] = (bf16_t)f1.y;
    h.h[6] = (bf16_t)f1.z; h.h[7] = (bf16_t)f1.w;
    *(uint4*)(Y + i) = h.u[0];
}

// ---------------------------------------------------------------------------
// Row softmax of P (axis=1) -> bf16 W[r][k]
// ---------------------------------------------------------------------------
__global__ __launch_bounds__(256)
void softmax_rows_kernel(const float* __restrict__ P, bf16_t* __restrict__ W)
{
    __shared__ float red[256];
    const int r = blockIdx.x, t = threadIdx.x;
    const float* row = P + (size_t)r * D_DIM;

    float v[4];
    #pragma unroll
    for (int q = 0; q < 4; ++q) v[q] = row[t + q * 256];

    float mx = fmaxf(fmaxf(v[0], v[1]), fmaxf(v[2], v[3]));
    red[t] = mx; __syncthreads();
    for (int s = 128; s > 0; s >>= 1) {
        if (t < s) red[t] = fmaxf(red[t], red[t + s]);
        __syncthreads();
    }
    mx = red[0]; __syncthreads();

    float e[4]; float sum = 0.f;
    #pragma unroll
    for (int q = 0; q < 4; ++q) { e[q] = __expf(v[q] - mx); sum += e[q]; }
    red[t] = sum; __syncthreads();
    for (int s = 128; s > 0; s >>= 1) {
        if (t < s) red[t] += red[t + s];
        __syncthreads();
    }
    const float inv = 1.0f / red[0];

    bf16_t* out = W + (size_t)r * D_DIM;
    #pragma unroll
    for (int q = 0; q < 4; ++q) out[t + q * 256] = (bf16_t)(e[q] * inv);
}

// ---------------------------------------------------------------------------
// s = softmax(P[c,:]);  W[c, n*32+i] = sum_o K[n,i,o] * s[n*32+o]
// ---------------------------------------------------------------------------
__global__ __launch_bounds__(256)
void softmax_block_kernel(const float* __restrict__ P,
                          const float* __restrict__ Kb,
                          bf16_t* __restrict__ W)
{
    __shared__ float sm[D_DIM];
    __shared__ float red[256];
    const int c = blockIdx.x, t = threadIdx.x;
    const float* row = P + (size_t)c * D_DIM;

    float v[4];
    #pragma unroll
    for (int q = 0; q < 4; ++q) v[q] = row[t + q * 256];

    float mx = fmaxf(fmaxf(v[0], v[1]), fmaxf(v[2], v[3]));
    red[t] = mx; __syncthreads();
    for (int s = 128; s > 0; s >>= 1) {
        if (t < s) red[t] = fmaxf(red[t], red[t + s]);
        __syncthreads();
    }
    mx = red[0]; __syncthreads();

    float e[4]; float sum = 0.f;
    #pragma unroll
    for (int q = 0; q < 4; ++q) { e[q] = __expf(v[q] - mx); sum += e[q]; }
    red[t] = sum; __syncthreads();
    for (int s = 128; s > 0; s >>= 1) {
        if (t < s) red[t] += red[t + s];
        __syncthreads();
    }
    const float inv = 1.0f / red[0];
    #pragma unroll
    for (int q = 0; q < 4; ++q) sm[t + q * 256] = e[q] * inv;
    __syncthreads();

    bf16_t* out = W + (size_t)c * D_DIM;
    #pragma unroll
    for (int q = 0; q < 4; ++q) {
        const int j = t + q * 256;
        const int n = j >> 5, i = j & 31;
        const float* kp = Kb + (size_t)((n << 5) + i) * 32;
        const float* sp = sm + (n << 5);
        float acc = 0.f;
        #pragma unroll
        for (int o = 0; o < 32; ++o) acc += kp[o] * sp[o];
        out[j] = (bf16_t)acc;
    }
}

// ---------------------------------------------------------------------------
// C[M,N] = A[M,K] @ W^T ; A bf16 [M][K], W bf16 [N][K].
// 256 threads = 8 waves, 128x256 block tile, 64x64 per wave (4x4 WMMA tiles),
// BK=32, double-buffered LDS fed by async global->LDS DMA.
// FINAL: fp32 out + bias; else bf16 out.
// ---------------------------------------------------------------------------
template <bool FINAL>
__global__ __launch_bounds__(256)
void gemm_bf16_kernel(const bf16_t* __restrict__ A,
                      const bf16_t* __restrict__ W,
                      void* __restrict__ Cptr,
                      const float* __restrict__ bias)
{
    __shared__ bf16_t As[2][BM * LDS_LD];   // 2*128*40*2 = 20 KB
    __shared__ bf16_t Bs[2][BN * LDS_LD];   // 2*256*40*2 = 40 KB

    const int tid  = threadIdx.x;
    const int lane = tid & 31;
    const int wid  = tid >> 5;
    const int wm   = (wid & 1) << 6;   // wave M offset: 0 / 64
    const int wn   = (wid >> 1) << 6;  // wave N offset: 0 / 64 / 128 / 192
    const int idx  = lane & 15;        // row/col within 16x16 tile
    const int half = lane >> 4;        // K-half / M-half selector

    const int m0 = blockIdx.y * BM;
    const int n0 = blockIdx.x * BN;

    v8f acc[4][4];
    #pragma unroll
    for (int i = 0; i < 4; ++i)
        #pragma unroll
        for (int j = 0; j < 4; ++j)
            acc[i][j] = v8f{0.f, 0.f, 0.f, 0.f, 0.f, 0.f, 0.f, 0.f};

    // Issue DMA for one K-step: A tile 8KB (2 segs/thread), B tile 16KB (4/thread).
    auto issue_tile = [&](int buf, int kt) {
        const int k0 = kt * BK;
        #pragma unroll
        for (int q = 0; q < 2; ++q) {
            const int s   = tid + q * 256;
            const int row = s >> 2;          // 0..127
            const int ce  = (s & 3) << 3;    // element offset 0/8/16/24
            copy16_g2l(A + (size_t)(m0 + row) * D_DIM + k0 + ce,
                       &As[buf][row * LDS_LD + ce]);
        }
        #pragma unroll
        for (int q = 0; q < 4; ++q) {
            const int s   = tid + q * 256;
            const int row = s >> 2;          // 0..255
            const int ce  = (s & 3) << 3;
            copy16_g2l(W + (size_t)(n0 + row) * D_DIM + k0 + ce,
                       &Bs[buf][row * LDS_LD + ce]);
        }
    };

    auto compute = [&](int buf) {
        // wave32 16-bit fragment layout: lane%16 = row, lane/16 selects K-half.
        Frag16 a[4];
        #pragma unroll
        for (int ms = 0; ms < 4; ++ms) {
            const bf16_t* p = &As[buf][(wm + ms * 16 + idx) * LDS_LD];
            a[ms].u[0] = *(const uint4*)(p + half * 8);
            a[ms].u[1] = *(const uint4*)(p + 16 + half * 8);
        }
        #pragma unroll
        for (int ns = 0; ns < 4; ++ns) {
            Frag16 b;
            const bf16_t* p = &Bs[buf][(wn + ns * 16 + idx) * LDS_LD];
            b.u[0] = *(const uint4*)(p + half * 8);
            b.u[1] = *(const uint4*)(p + 16 + half * 8);
            #pragma unroll
            for (int ms = 0; ms < 4; ++ms)
                acc[ms][ns] = __builtin_amdgcn_wmma_f32_16x16x32_bf16(
                    false, a[ms].v, false, b.v,
                    (short)0, acc[ms][ns], false, false);
        }
    };

    issue_tile(0, 0);
    wait_async_zero();
    __syncthreads();

    for (int kt = 0; kt < NK; ++kt) {
        const int cur = kt & 1;
        // Overwrite the other buffer: its last readers finished before the
        // barrier that ended iteration kt-1.
        if (kt + 1 < NK) issue_tile(cur ^ 1, kt + 1);
        compute(cur);
        if (kt + 1 < NK) wait_async_zero();  // my DMAs for kt+1 have landed
        __syncthreads();                     // everyone done reading cur + all DMAs visible
    }

    // C/D layout: VGPR r -> M = r + 8*(lane/16); N = lane%16
    #pragma unroll
    for (int ms = 0; ms < 4; ++ms) {
        const int row_base = m0 + wm + ms * 16 + half * 8;
        #pragma unroll
        for (int ns = 0; ns < 4; ++ns) {
            const int col = n0 + wn + ns * 16 + idx;
            #pragma unroll
            for (int r = 0; r < 8; ++r) {
                const size_t o = (size_t)(row_base + r) * D_DIM + col;
                if constexpr (FINAL) {
                    ((float*)Cptr)[o] = acc[ms][ns][r] + bias[col];
                } else {
                    ((bf16_t*)Cptr)[o] = (bf16_t)acc[ms][ns][r];
                }
            }
        }
    }
}

// ---------------------------------------------------------------------------
extern "C" void kernel_launch(void* const* d_in, const int* in_sizes, int n_in,
                              void* d_out, int out_size, void* d_ws, size_t ws_size,
                              hipStream_t stream)
{
    (void)in_sizes; (void)n_in; (void)out_size; (void)ws_size;

    const float* x    = (const float*)d_in[0];
    const float* P0   = (const float*)d_in[1];
    const float* P1   = (const float*)d_in[2];
    const float* P2   = (const float*)d_in[3];
    const float* P3   = (const float*)d_in[4];
    const float* K0   = (const float*)d_in[5];
    const float* K1   = (const float*)d_in[6];
    const float* K2   = (const float*)d_in[7];
    const float* bias = (const float*)d_in[8];

    const size_t MAT = (size_t)D_DIM * D_DIM;
    const size_t YN  = (size_t)B_DIM * D_DIM;

    bf16_t* W0 = (bf16_t*)d_ws;
    bf16_t* W1 = W0 + MAT;
    bf16_t* W2 = W1 + MAT;
    bf16_t* W3 = W2 + MAT;
    bf16_t* Ya = W3 + MAT;                   // 67 MB intermediate in workspace
    bf16_t* Xc = (bf16_t*)d_out;             // bf16 x in low half of d_out
    bf16_t* Yb = (bf16_t*)d_out + YN;        // intermediate in high half of d_out

    // Precompute: bf16 x, and bf16 N-major weights (softmax + folded blockdiag).
    convert_bf16_kernel <<<(YN / (256 * 8)), 256, 0, stream>>>(x, Xc);
    softmax_rows_kernel <<<D_DIM, 256, 0, stream>>>(P0, W0);
    softmax_block_kernel<<<D_DIM, 256, 0, stream>>>(P1, K0, W1);
    softmax_block_kernel<<<D_DIM, 256, 0, stream>>>(P2, K1, W2);
    softmax_block_kernel<<<D_DIM, 256, 0, stream>>>(P3, K2, W3);

    // Four chained WMMA GEMMs.  Final one reads only ws and fully rewrites d_out.
    dim3 grid(D_DIM / BN, B_DIM / BM);
    dim3 blk(256);
    gemm_bf16_kernel<false><<<grid, blk, 0, stream>>>(Xc, W0, Ya, nullptr);
    gemm_bf16_kernel<false><<<grid, blk, 0, stream>>>(Ya, W1, Yb, nullptr);
    gemm_bf16_kernel<false><<<grid, blk, 0, stream>>>(Yb, W2, Ya, nullptr);
    gemm_bf16_kernel<true ><<<grid, blk, 0, stream>>>(Ya, W3, d_out, bias);
}
